// GraphTripletConvModule_63007170232987
// MI455X (gfx1250) — compile-verified, matrix operand order checked by reading, not run on hardware
//
#include <hip/hip_runtime.h>
#include <hip/hip_bf16.h>

#define N_NODES 10000
#define N_EDGES 320000
#define DN 256   // node feature dim
#define DE 32    // edge input dim

typedef __bf16 v16bf __attribute__((ext_vector_type(16)));
typedef __bf16 v8bf  __attribute__((ext_vector_type(8)));
typedef float  v8f   __attribute__((ext_vector_type(8)));

// ---------------------------------------------------------------------------
// WMMA helper: D = A(16x32 bf16) * B(32x16 bf16) + C(16x16 f32)
// ---------------------------------------------------------------------------
static __device__ __forceinline__ v8f wmma_bf16(v16bf a, v16bf b, v8f c) {
  return __builtin_amdgcn_wmma_f32_16x16x32_bf16(false, a, false, b, (short)0, c,
                                                 false, false);
}

// A-fragment loader from a per-lane row pointer (row-major bf16 rows).
// Per CDNA5 ISA 16-bit A layout: lanes 0-15 -> M=lane, K groups {0..7,16..23};
// lanes 16-31 -> M=lane-16, K groups {8..15,24..31}. kbase = (lane>>4)*8.
static __device__ __forceinline__ v16bf load_a_bf16(const __bf16* __restrict__ row,
                                                    int koff, int kbase) {
  v8bf lo = *(const v8bf*)(row + koff + kbase);        // 16B contiguous
  v8bf hi = *(const v8bf*)(row + koff + kbase + 16);   // 16B contiguous
  v16bf a;
#pragma unroll
  for (int i = 0; i < 8; ++i) { a[i] = lo[i]; a[i + 8] = hi[i]; }
  return a;
}

// Same, but converting from an f32 source row on the fly.
static __device__ __forceinline__ v16bf load_a_f32(const float* __restrict__ row,
                                                   int koff, int kbase) {
  v16bf a;
#pragma unroll
  for (int i = 0; i < 8; ++i) {
    a[i]     = (__bf16)row[koff + kbase + i];
    a[i + 8] = (__bf16)row[koff + kbase + 16 + i];
  }
  return a;
}

// B-fragment loader: weights pre-packed so each lane reads 16 contiguous bf16.
static __device__ __forceinline__ v16bf load_b(const __bf16* __restrict__ packed,
                                               int frag, int lane) {
  return *(const v16bf*)(packed + (size_t)frag * 512 + lane * 16);
}

// ---------------------------------------------------------------------------
// Pack a row-major f32 weight [K x Nn] into bf16 WMMA B fragments.
// frag f = ktile*(Nn/16)+ntile; element (lane,i): n = nt*16 + (lane&15),
// k = kt*32 + (lane>>4)*8 + (i<8 ? i : 16+i-8). Matches load_b + ISA B layout.
// ---------------------------------------------------------------------------
__global__ void pack_w_kernel(const float* __restrict__ W, __bf16* __restrict__ out,
                              int K, int Nn) {
  int tid = blockIdx.x * blockDim.x + threadIdx.x;
  if (tid >= K * Nn) return;
  int i    = tid & 15;
  int lane = (tid >> 4) & 31;
  int frag = tid >> 9;
  int ntiles = Nn >> 4;
  int nt = frag % ntiles;
  int kt = frag / ntiles;
  int n  = nt * 16 + (lane & 15);
  int k  = kt * 32 + ((lane >> 4) * 8) + ((i < 8) ? i : (16 + (i - 8)));
  out[tid] = (__bf16)W[k * Nn + n];
}

// ---------------------------------------------------------------------------
// nodenet: nf = relu(input @ Wn + bn), stored bf16 row-major [N_NODES, 256].
// Block = 256 thr (8 waves) handles 16 rows x 256 cols; wave w -> ntiles 2w,2w+1.
// ---------------------------------------------------------------------------
__global__ __launch_bounds__(256) void nodenet_kernel(
    const float* __restrict__ input, const __bf16* __restrict__ pWn,
    const float* __restrict__ bn, __bf16* __restrict__ nf) {
  const int lane  = threadIdx.x & 31;
  const int wave  = threadIdx.x >> 5;
  const int row0  = blockIdx.x * 16;
  const int m     = lane & 15;
  const int kbase = (lane >> 4) * 8;
  const int nt0 = 2 * wave, nt1 = 2 * wave + 1;

  const float* arow = input + (size_t)(row0 + m) * DN;
  v8f c0 = {}, c1 = {};
#pragma unroll
  for (int kt = 0; kt < 8; ++kt) {
    v16bf a  = load_a_f32(arow, kt * 32, kbase);
    v16bf b0 = load_b(pWn, kt * 16 + nt0, lane);
    v16bf b1 = load_b(pWn, kt * 16 + nt1, lane);
    c0 = wmma_bf16(a, b0, c0);
    c1 = wmma_bf16(a, b1, c1);
  }
  // C/D layout: lane -> col n = (lane&15); rows m = (lane>=16?8:0)+r
  const int nA = nt0 * 16 + (lane & 15);
  const int nB = nt1 * 16 + (lane & 15);
  const int mb = (lane >> 4) * 8;
  const float bA = bn[nA], bB = bn[nB];
#pragma unroll
  for (int r = 0; r < 8; ++r) {
    float vA = fmaxf(c0[r] + bA, 0.f);
    float vB = fmaxf(c1[r] + bB, 0.f);
    nf[(size_t)(row0 + mb + r) * DN + nA] = (__bf16)vA;
    nf[(size_t)(row0 + mb + r) * DN + nB] = (__bf16)vB;
  }
}

// ---------------------------------------------------------------------------
// Fused edgenet + gather + mixnet + segment-sum scatter.
// Block = 256 thr (8 waves) handles 64 edges x all 256 out channels.
// Wave w: ntiles {2w,2w+1} x 4 mtiles -> 8 f32 accumulators; B frags are
// unique per wave (no duplication); A gathers hit WGP$/L2 (nf is 5MB bf16).
// ---------------------------------------------------------------------------
__global__ __launch_bounds__(256) void mixnet_kernel(
    const float* __restrict__ edgefeats, const int* __restrict__ idxn,
    const int* __restrict__ seg_ids, const __bf16* __restrict__ nf,
    const __bf16* __restrict__ pWe, const float* __restrict__ be,
    const __bf16* __restrict__ pWm, const float* __restrict__ bm,
    float* __restrict__ acc) {
  __shared__ __bf16 ef_lds[64 * DN];   // relu(edgenet) tile, 32 KB
  __shared__ int idx_lds[64];
  __shared__ int seg_lds[64];

  const int lane  = threadIdx.x & 31;
  const int wave  = threadIdx.x >> 5;
  const int e0    = blockIdx.x * 64;
  const int m     = lane & 15;
  const int kbase = (lane >> 4) * 8;
  const int nt0 = 2 * wave, nt1 = 2 * wave + 1;

  if (threadIdx.x < 64) {
    idx_lds[threadIdx.x] = idxn[e0 + threadIdx.x];
    seg_lds[threadIdx.x] = seg_ids[e0 + threadIdx.x];
  }

  // ---- Phase 1: ef = relu(edgefeats @ We + be) for this block's 64 edges ----
  {
    v8f c[4][2] = {};
    v16bf b0 = load_b(pWe, nt0, lane);   // We: single K-tile (K=32)
    v16bf b1 = load_b(pWe, nt1, lane);
#pragma unroll
    for (int mt = 0; mt < 4; ++mt) {
      const float* ar = edgefeats + (size_t)(e0 + mt * 16 + m) * DE;
      v16bf a = load_a_f32(ar, 0, kbase);
      c[mt][0] = wmma_bf16(a, b0, c[mt][0]);
      c[mt][1] = wmma_bf16(a, b1, c[mt][1]);
    }
    const int nA = nt0 * 16 + (lane & 15);
    const int nB = nt1 * 16 + (lane & 15);
    const int mb = (lane >> 4) * 8;
    const float bA = be[nA], bB = be[nB];
#pragma unroll
    for (int mt = 0; mt < 4; ++mt) {
#pragma unroll
      for (int r = 0; r < 8; ++r) {
        float vA = fmaxf(c[mt][0][r] + bA, 0.f);
        float vB = fmaxf(c[mt][1][r] + bB, 0.f);
        ef_lds[(mt * 16 + mb + r) * DN + nA] = (__bf16)vA;
        ef_lds[(mt * 16 + mb + r) * DN + nB] = (__bf16)vB;
      }
    }
  }
  __syncthreads();

  // ---- Phase 2: out = relu([nf[idxn] | nf[seg] | ef] @ Wm + bm) ----
  v8f d[4][2] = {};
  const __bf16* rowp[2][4];
#pragma unroll
  for (int mt = 0; mt < 4; ++mt) {
    rowp[0][mt] = nf + (size_t)idx_lds[mt * 16 + m] * DN;
    rowp[1][mt] = nf + (size_t)seg_lds[mt * 16 + m] * DN;
  }
  // K-tiles 0..15: gathered node features (global, bf16, b128 loads)
#pragma unroll
  for (int src = 0; src < 2; ++src) {
#pragma unroll
    for (int kt = 0; kt < 8; ++kt) {
      v16bf b0 = load_b(pWm, (src * 8 + kt) * 16 + nt0, lane);
      v16bf b1 = load_b(pWm, (src * 8 + kt) * 16 + nt1, lane);
#pragma unroll
      for (int mt = 0; mt < 4; ++mt) {
        v16bf a = load_a_bf16(rowp[src][mt], kt * 32, kbase);
        d[mt][0] = wmma_bf16(a, b0, d[mt][0]);
        d[mt][1] = wmma_bf16(a, b1, d[mt][1]);
      }
    }
  }
  // K-tiles 16..23: ef tile from LDS
#pragma unroll
  for (int kt = 0; kt < 8; ++kt) {
    v16bf b0 = load_b(pWm, (16 + kt) * 16 + nt0, lane);
    v16bf b1 = load_b(pWm, (16 + kt) * 16 + nt1, lane);
#pragma unroll
    for (int mt = 0; mt < 4; ++mt) {
      const int base = (mt * 16 + m) * DN + kt * 32 + kbase;
      v16bf a;
#pragma unroll
      for (int i = 0; i < 8; ++i) {
        a[i]     = ef_lds[base + i];
        a[i + 8] = ef_lds[base + 16 + i];
      }
      d[mt][0] = wmma_bf16(a, b0, d[mt][0]);
      d[mt][1] = wmma_bf16(a, b1, d[mt][1]);
    }
  }

  // ---- Epilogue: bias + relu + segment-sum scatter (acc is L2-resident) ----
  const int nA = nt0 * 16 + (lane & 15);
  const int nB = nt1 * 16 + (lane & 15);
  const int mb = (lane >> 4) * 8;
  const float bA = bm[nA], bB = bm[nB];
#pragma unroll
  for (int mt = 0; mt < 4; ++mt) {
#pragma unroll
    for (int r = 0; r < 8; ++r) {
      const int s = seg_lds[mt * 16 + mb + r];
      float vA = fmaxf(d[mt][0][r] + bA, 0.f);
      float vB = fmaxf(d[mt][1][r] + bB, 0.f);
      atomicAdd(&acc[(size_t)s * DN + nA], vA);
      atomicAdd(&acc[(size_t)s * DN + nB], vB);
    }
  }
}

// ---------------------------------------------------------------------------
// Small epilogue/prologue kernels
// ---------------------------------------------------------------------------
__global__ void zero_kernel(float* __restrict__ acc, float* __restrict__ cnt) {
  int i = blockIdx.x * blockDim.x + threadIdx.x;
  if (i < N_NODES * DN) acc[i] = 0.f;
  if (i < N_NODES) cnt[i] = 0.f;
}

__global__ void count_kernel(const int* __restrict__ seg, float* __restrict__ cnt) {
  int i = blockIdx.x * blockDim.x + threadIdx.x;
  if (i < N_EDGES) atomicAdd(&cnt[seg[i]], 1.0f);
}

__global__ void finalize_kernel(const float* __restrict__ acc,
                                const float* __restrict__ cnt,
                                float* __restrict__ out) {
  int i = blockIdx.x * blockDim.x + threadIdx.x;
  if (i >= N_NODES * DN) return;
  float c = cnt[i >> 8];
  out[i] = (c > 0.f) ? acc[i] / c : 0.f;
}

// ---------------------------------------------------------------------------
// Launch
// ---------------------------------------------------------------------------
extern "C" void kernel_launch(void* const* d_in, const int* in_sizes, int n_in,
                              void* d_out, int out_size, void* d_ws, size_t ws_size,
                              hipStream_t stream) {
  const float* input     = (const float*)d_in[0];
  const float* edgefeats = (const float*)d_in[1];
  const int*   idxn      = (const int*)d_in[2];
  const int*   seg_ids   = (const int*)d_in[3];
  const float* Wn        = (const float*)d_in[4];
  const float* bn        = (const float*)d_in[5];
  const float* We        = (const float*)d_in[6];
  const float* be        = (const float*)d_in[7];
  const float* Wm        = (const float*)d_in[8];
  const float* bm        = (const float*)d_in[9];
  float* out = (float*)d_out;

  // Workspace layout (all 256B aligned):
  char* ws = (char*)d_ws;
  __bf16* pWn = (__bf16*)(ws);                         // 256*256*2   = 128 KB
  __bf16* pWe = (__bf16*)(ws + 131072);                // 32*256*2    = 16 KB
  __bf16* pWm = (__bf16*)(ws + 147456);                // 768*256*2   = 384 KB
  __bf16* nf  = (__bf16*)(ws + 540672);                // 10000*256*2 = 5.0 MB
  float*  acc = (float*)(ws + 5660672);                // 10000*256*4 = 10.0 MB
  float*  cnt = (float*)(ws + 15900672);               // 10000*4     = 40 KB

  zero_kernel<<<(N_NODES * DN + 255) / 256, 256, 0, stream>>>(acc, cnt);

  pack_w_kernel<<<(256 * 256 + 255) / 256, 256, 0, stream>>>(Wn, pWn, 256, 256);
  pack_w_kernel<<<(32 * 256 + 255) / 256, 256, 0, stream>>>(We, pWe, 32, 256);
  pack_w_kernel<<<(768 * 256 + 255) / 256, 256, 0, stream>>>(Wm, pWm, 768, 256);

  nodenet_kernel<<<N_NODES / 16, 256, 0, stream>>>(input, pWn, bn, nf);

  count_kernel<<<(N_EDGES + 255) / 256, 256, 0, stream>>>(seg_ids, cnt);

  mixnet_kernel<<<N_EDGES / 64, 256, 0, stream>>>(edgefeats, idxn, seg_ids, nf,
                                                  pWe, be, pWm, bm, acc);

  finalize_kernel<<<(N_NODES * DN + 255) / 256, 256, 0, stream>>>(acc, cnt, out);
}
